// ES_MOE_65180423685431
// MI455X (gfx1250) — matrix operand order repaired and checked by reading, not compiled, not run on hardware
//
#include <hip/hip_runtime.h>

typedef __attribute__((ext_vector_type(16))) _Float16 v16h;
typedef __attribute__((ext_vector_type(8)))  float    v8f;

#define Bn   16
#define Cn   256
#define Hn   64
#define Wn   64
#define En   3
#define REDn 32
#define THRv 0.4f
#define BN_EPS 1e-5f
#define HWn  4096
#define KCH  8        // K chunks of 32 channels
#define NT   16       // N tiles of 16 channels

// ---------------- Kernel 0: pre-pack pointwise weights into f16 B-fragment layout
// bpack fragment index: ((e*KCH + kc)*NT + ntile)*32 + lane  -> 16 halfs (32B)
// lane holds N = ntile*16 + (lane&15); VGPR v holds K = kc*32 + kb + {2v, 2v+1},
// kb = (lane<16) ? 0 : 16   (contiguous K order; A in k_moe uses the same)
__global__ __launch_bounds__(256) void k_packw(
    const float* __restrict__ pw0, const float* __restrict__ pw1,
    const float* __restrict__ pw2, v16h* __restrict__ bpack)
{
    int idx = blockIdx.x * 256 + threadIdx.x;   // 0 .. 3*8*16*32-1 = 12287
    int lane  = idx & 31;
    int ntile = (idx >> 5) & 15;
    int kc    = (idx >> 9) & 7;
    int e     = idx >> 12;
    const float* pwe = (e == 0) ? pw0 : (e == 1 ? pw1 : pw2);
    int n  = ntile * 16 + (lane & 15);
    int kb = (lane < 16) ? 0 : 16;
    const float* src = pwe + n * Cn + kc * 32 + kb;
    v16h f;
    #pragma unroll
    for (int v = 0; v < 8; ++v) {
        f[2 * v]     = (_Float16)src[2 * v];
        f[2 * v + 1] = (_Float16)src[2 * v + 1];
    }
    bpack[idx] = f;
}

// ---------------- Kernel 1: routing MLP + softmax (per pixel) ----------------
__global__ __launch_bounds__(256) void k_routing(
    const float* __restrict__ x,
    const float* __restrict__ rw1, const float* __restrict__ rb1,
    const float* __restrict__ rw2, const float* __restrict__ rb2,
    float* __restrict__ r_out)
{
    int pixel = blockIdx.x * 256 + threadIdx.x;      // 0..65535
    int b  = pixel >> 12;
    int pp = pixel & 4095;

    float h[REDn];
    #pragma unroll
    for (int rr = 0; rr < REDn; ++rr) h[rr] = rb1[rr];

    for (int c = 0; c < Cn; ++c) {
        float xv = x[(b * Cn + c) * HWn + pp];
        #pragma unroll
        for (int rr = 0; rr < REDn; ++rr) h[rr] = fmaf(rw1[rr * Cn + c], xv, h[rr]);
    }
    #pragma unroll
    for (int rr = 0; rr < REDn; ++rr) h[rr] = fmaxf(h[rr], 0.0f);

    float lg[En];
    #pragma unroll
    for (int e = 0; e < En; ++e) {
        float a = rb2[e];
        #pragma unroll
        for (int rr = 0; rr < REDn; ++rr) a = fmaf(rw2[e * REDn + rr], h[rr], a);
        lg[e] = a;
    }
    float m = fmaxf(lg[0], fmaxf(lg[1], lg[2]));
    float s = 0.0f;
    #pragma unroll
    for (int e = 0; e < En; ++e) { lg[e] = expf(lg[e] - m); s += lg[e]; }
    float inv = 1.0f / s;
    #pragma unroll
    for (int e = 0; e < En; ++e)
        r_out[(b * En + e) * HWn + pp] = lg[e] * inv;
}

// ------------- Kernel 2: importance reduction + top-k gating (per batch) -----
__global__ __launch_bounds__(256) void k_select(
    const float* __restrict__ r, float* __restrict__ sel)
{
    __shared__ float red[256];
    __shared__ float imp[En];
    int b = blockIdx.x;
    int t = threadIdx.x;

    for (int e = 0; e < En; ++e) {
        float s = 0.0f;
        for (int i = t; i < HWn; i += 256) s += r[(b * En + e) * HWn + i];
        red[t] = s;
        __syncthreads();
        for (int off = 128; off > 0; off >>= 1) {
            if (t < off) red[t] += red[t + off];
            __syncthreads();
        }
        if (t == 0) imp[e] = red[0] * (1.0f / (float)HWn);
        __syncthreads();
    }
    if (t == 0) {
        int i1 = 0;
        for (int e = 1; e < En; ++e) if (imp[e] > imp[i1]) i1 = e;     // top-1 (first on tie)
        int i2 = -1;
        for (int e = 0; e < En; ++e)
            if (e != i1 && (i2 < 0 || imp[e] > imp[i2])) i2 = e;       // top-2
        for (int e = 0; e < En; ++e) {
            float keep = (e == i1) ? 1.0f
                       : ((e == i2 && imp[i2] >= THRv) ? 1.0f : 0.0f);
            sel[b * En + e] = keep;
        }
    }
}

// ------------- Kernel 3: fused dwconv -> WMMA pointwise -> gate -> BN+SiLU ---
// One block per 16-pixel row segment. 8 waves; wave w handles N-tiles 2w, 2w+1.
__global__ __launch_bounds__(256) void k_moe(
    const float* __restrict__ x,
    const float* __restrict__ dw0, const float* __restrict__ dw1, const float* __restrict__ dw2,
    const v16h* __restrict__ bpack,
    const float* __restrict__ pb0, const float* __restrict__ pb1, const float* __restrict__ pb2,
    const float* __restrict__ bn_g, const float* __restrict__ bn_b,
    const float* __restrict__ bn_m, const float* __restrict__ bn_v,
    const float* __restrict__ r_in, const float* __restrict__ sel_in,
    float* __restrict__ out)
{
    __shared__ float    xs[32][7][22];       // 32 ch x (7 rows, 22 cols with 3-halo)
    __shared__ _Float16 As[En][16][32];      // A fragments: [expert][pixel M][K]
    __shared__ float    rs[En][16];          // per-pixel routing weights for tile
    __shared__ float    sel_s[En];           // per-batch expert keep {0,1}
    __shared__ float    outs[Cn * 16];       // transpose buffer for coalesced store

    const int tid   = threadIdx.x;
    const int lane  = tid & 31;
    const int wave  = tid >> 5;
    const int col   = lane & 15;             // N (and pixel M) lane column
    const int halfw = lane >> 4;             // 0: lanes 0-15, 1: lanes 16-31

    const int id  = blockIdx.x;              // B*H*W/16 = 4096 blocks
    const int b   = id >> 8;
    const int rem = id & 255;
    const int y   = rem >> 2;
    const int x0  = (rem & 3) << 4;

    // stage routing weights + gating scales
    if (tid < En * 16) {
        int e = tid >> 4, p = tid & 15;
        rs[e][p] = r_in[(b * En + e) * HWn + y * Wn + x0 + p];
    }
    if (tid >= 64 && tid < 64 + En) sel_s[tid - 64] = sel_in[b * En + (tid - 64)];

    v8f acc[En][2];
    #pragma unroll
    for (int e = 0; e < En; ++e)
        #pragma unroll
        for (int t = 0; t < 2; ++t) { v8f z = {}; acc[e][t] = z; }

    const int ntile0 = wave * 2;

    for (int kc = 0; kc < KCH; ++kc) {       // K chunks of 32 channels
        __syncthreads();
        // ---- stage x chunk (32 ch, 7x22 halo tile), zero-pad SAME borders ----
        for (int idx = tid; idx < 32 * 7 * 22; idx += 256) {
            int k    = idx / 154;
            int r2   = idx - k * 154;
            int row  = r2 / 22;
            int colx = r2 - row * 22;
            int yy = y + row - 3;
            int xx = x0 + colx - 3;
            float v = 0.0f;
            if ((unsigned)yy < (unsigned)Hn && (unsigned)xx < (unsigned)Wn)
                v = x[((b * Cn + kc * 32 + k) * Hn + yy) * Wn + xx];
            xs[k][row][colx] = v;
        }
        if (kc < KCH - 1)   // hint next chunk into cache (global_prefetch_b8)
            __builtin_prefetch(&x[((b * Cn + (kc + 1) * 32) * Hn + y) * Wn + x0] + tid, 0, 0);
        __syncthreads();

        // ---- cooperative depthwise conv -> f16 A fragments in LDS ----
        // 3 experts x 32 K x 16 M = 1536 values, 6 per thread
        #pragma unroll
        for (int it = 0; it < 6; ++it) {
            int idx = tid + it * 256;        // 0..1535
            int e  = idx >> 9;
            int r2 = idx & 511;
            int k  = r2 >> 4;
            int p  = r2 & 15;
            int ksz = 3 + 2 * e;             // 3,5,7
            int rad = 1 + e;
            const float* wk = (e == 0 ? dw0 : (e == 1 ? dw1 : dw2))
                              + (kc * 32 + k) * ksz * ksz;
            float a = 0.0f;
            for (int i = 0; i < ksz; ++i)
                for (int j = 0; j < ksz; ++j)
                    a = fmaf(xs[k][3 - rad + i][p + 3 - rad + j], wk[i * ksz + j], a);
            As[e][p][k] = (_Float16)a;
        }
        __syncthreads();

        // ---- per-wave WMMA: acc[e][t] += A_e x Wpw_e  (16x16x32, f32 acc) ----
        const int kbaseA = halfw ? 16 : 0;   // contiguous K layout (matches bpack)
        #pragma unroll
        for (int e = 0; e < En; ++e) {
            const _Float16* ap = &As[e][col][0];
            v16h afrag;
            #pragma unroll
            for (int v = 0; v < 8; ++v) {
                int k = kbaseA + 2 * v;
                afrag[2 * v]     = ap[k];
                afrag[2 * v + 1] = ap[k + 1];
            }
            #pragma unroll
            for (int t = 0; t < 2; ++t) {
                // pre-packed, pre-swizzled f16 B fragment: one coalesced 32B load
                v16h bfrag = bpack[((e * KCH + kc) * NT + (ntile0 + t)) * 32 + lane];
                acc[e][t] = __builtin_amdgcn_wmma_f32_16x16x32_f16(
                    false, afrag, false, bfrag, (short)0, acc[e][t], false, false);
            }
        }
    }

    __syncthreads();
    // ---- epilogue: gate, combine experts, BN + SiLU, transpose in LDS ----
    #pragma unroll
    for (int t = 0; t < 2; ++t) {
        int n = (ntile0 + t) * 16 + col;
        float sc = bn_g[n] * rsqrtf(bn_v[n] + BN_EPS);
        float mu = bn_m[n];
        float bt = bn_b[n];
        float pb[En];
        #pragma unroll
        for (int e = 0; e < En; ++e)
            pb[e] = (e == 0 ? pb0 : (e == 1 ? pb1 : pb2))[n];
        #pragma unroll
        for (int j = 0; j < 8; ++j) {
            int p = halfw ? (j + 8) : j;     // D layout: lanes 0-15 M=j, 16-31 M=j+8
            float v = 0.0f;
            #pragma unroll
            for (int e = 0; e < En; ++e)
                v += (acc[e][t][j] + pb[e]) * rs[e][p] * sel_s[e];
            v = (v - mu) * sc + bt;
            v = v / (1.0f + __expf(-v));     // SiLU
            outs[n * 16 + p] = v;
        }
    }
    __syncthreads();
    // coalesced store: 16 consecutive pixels (64B) per channel row
    #pragma unroll
    for (int i = 0; i < 16; ++i) {
        int cch = i * 16 + (tid >> 4);
        int p   = tid & 15;
        out[((b * Cn + cch) * Hn + y) * Wn + x0 + p] = outs[cch * 16 + p];
    }
}

// ---------------------------------------------------------------------------
extern "C" void kernel_launch(void* const* d_in, const int* in_sizes, int n_in,
                              void* d_out, int out_size, void* d_ws, size_t ws_size,
                              hipStream_t stream)
{
    const float* x    = (const float*)d_in[0];
    const float* rw1  = (const float*)d_in[1];
    const float* rb1  = (const float*)d_in[2];
    const float* rw2  = (const float*)d_in[3];
    const float* rb2  = (const float*)d_in[4];
    const float* dw0  = (const float*)d_in[5];
    const float* dw1  = (const float*)d_in[6];
    const float* dw2  = (const float*)d_in[7];
    const float* pw0  = (const float*)d_in[8];
    const float* pw1  = (const float*)d_in[9];
    const float* pw2  = (const float*)d_in[10];
    const float* pb0  = (const float*)d_in[11];
    const float* pb1  = (const float*)d_in[12];
    const float* pb2  = (const float*)d_in[13];
    const float* bn_g = (const float*)d_in[14];
    const float* bn_b = (const float*)d_in[15];
    const float* bn_m = (const float*)d_in[16];
    const float* bn_v = (const float*)d_in[17];
    float* out = (float*)d_out;

    // workspace layout (32B-aligned base from the harness):
    //   bpack : 3*8*16*32 v16h fragments = 393,216 B
    //   r_ws  : B*E*H*W floats           = 786,432 B
    //   sel_ws: B*E floats
    v16h*  bpack  = (v16h*)d_ws;
    float* r_ws   = (float*)((char*)d_ws + (size_t)En * KCH * NT * 32 * sizeof(v16h));
    float* sel_ws = r_ws + (size_t)Bn * En * HWn;

    k_packw<<<(En * KCH * NT * 32) / 256, 256, 0, stream>>>(pw0, pw1, pw2, bpack);
    k_routing<<<(Bn * HWn) / 256, 256, 0, stream>>>(x, rw1, rb1, rw2, rb2, r_ws);
    k_select<<<Bn, 256, 0, stream>>>(r_ws, sel_ws);
    k_moe<<<(Bn * HWn) / 16, 256, 0, stream>>>(
        x, dw0, dw1, dw2, bpack, pb0, pb1, pb2,
        bn_g, bn_b, bn_m, bn_v, r_ws, sel_ws, out);
}
